// RecommenderRNN_67242007986320
// MI455X (gfx1250) — matrix operand order repaired and checked by reading, not compiled, not run on hardware
//
#include <hip/hip_runtime.h>

// Problem dims (fixed by reference)
#define TT 256
#define BB 256
#define HH 512

typedef __attribute__((ext_vector_type(16))) __bf16 v16bf;
typedef __attribute__((ext_vector_type(8)))  float  v8f;

struct Q2 { uint4 a, b; };   // 32 bytes == one v16bf fragment

// ---------- helpers ----------
__device__ __forceinline__ unsigned short f2bf(float f) {
  unsigned int u = __float_as_uint(f);
  u += 0x7FFFu + ((u >> 16) & 1u);     // round-to-nearest-even
  return (unsigned short)(u >> 16);
}
__device__ __forceinline__ unsigned int pk2bf(float lo, float hi) {
  return (unsigned int)f2bf(lo) | ((unsigned int)f2bf(hi) << 16);
}

__device__ __forceinline__ v8f wmma_bf16(v16bf a, v16bf b, v8f c) {
  // D = A(16x32) * B(32x16) + C, fp32 accum
  return __builtin_amdgcn_wmma_f32_16x16x32_bf16(false, a, false, b, (short)0, c, false, false);
}

// B fragment: B[k,n] = W[n,k], W row-major [N x 512] bf16.
// Lane layout (ISA 7.12.2): lane n=l15, chunks K=[k0+8h .. +7] and [k0+16+8h .. +7].
__device__ __forceinline__ v16bf load_b_frag(const unsigned short* __restrict__ W,
                                             int n0, int k0, int l15, int half) {
  const unsigned short* p = W + (size_t)(n0 + l15) * HH + k0 + 8 * half;
  Q2 q{*(const uint4*)p, *(const uint4*)(p + 16)};
  return __builtin_bit_cast(v16bf, q);
}

// A fragment from bf16 row-major: lane row m=l15, chunks K=[k0+8h .. +7] and [k0+16+8h .. +7]
__device__ __forceinline__ v16bf load_a_bf16(const unsigned short* __restrict__ row,
                                             int k0, int half) {
  Q2 q{*(const uint4*)(row + k0 + 8 * half),
       *(const uint4*)(row + k0 + 16 + 8 * half)};
  return __builtin_bit_cast(v16bf, q);
}

// A fragment from fp32 row (convert on the fly)
__device__ __forceinline__ v16bf load_a_f32(const float* __restrict__ row, int k0, int half) {
  const float4* p0 = (const float4*)(row + k0 + 8 * half);
  const float4* p1 = (const float4*)(row + k0 + 16 + 8 * half);
  float4 a = p0[0], b = p0[1], c = p1[0], d = p1[1];
  Q2 q{make_uint4(pk2bf(a.x, a.y), pk2bf(a.z, a.w), pk2bf(b.x, b.y), pk2bf(b.z, b.w)),
       make_uint4(pk2bf(c.x, c.y), pk2bf(c.z, c.w), pk2bf(d.x, d.y), pk2bf(d.z, d.w))};
  return __builtin_bit_cast(v16bf, q);
}

// ---------- weight fp32 -> bf16 ----------
__global__ void cvt_f32_to_bf16(const float* __restrict__ s,
                                unsigned short* __restrict__ d, int n) {
  int i = blockIdx.x * blockDim.x + threadIdx.x;
  if (i < n) d[i] = f2bf(s[i]);
}

// ---------- XW = A @ Wih^T + (bih + bhh), M = T*B = 65536, N = K = 512 ----------
// Block: 8 waves (4 row x 2 col), wave tile 32x32. Grid (512, 8).
template <bool LAYER0>
__global__ __launch_bounds__(256) void xw_gemm(const void* __restrict__ Ain,
                                               const unsigned short* __restrict__ Wih,
                                               const float* __restrict__ bih,
                                               const float* __restrict__ bhh,
                                               float* __restrict__ XW) {
  const int wave = threadIdx.x >> 5;
  const int lane = threadIdx.x & 31;
  const int l15 = lane & 15, half = lane >> 4;
  const int r0 = blockIdx.x * 128 + (wave & 3) * 32;
  const int n0 = blockIdx.y * 64 + (wave >> 2) * 32;

  v8f acc[2][2] = {};
  for (int ki = 0; ki < 16; ++ki) {
    const int k0 = ki * 32;
    v16bf af[2], bf[2];
#pragma unroll
    for (int ar = 0; ar < 2; ++ar) {
      int r = r0 + ar * 16 + l15;          // row index in [T,B] flat order
      if (LAYER0) {
        int t = r >> 8, bb = r & 255;      // x is [B,T,I]
        const float* rp = (const float*)Ain + ((size_t)bb * TT + t) * HH;
        af[ar] = load_a_f32(rp, k0, half);
      } else {
        const unsigned short* rp = (const unsigned short*)Ain + (size_t)r * HH;
        af[ar] = load_a_bf16(rp, k0, half);
      }
    }
#pragma unroll
    for (int ac = 0; ac < 2; ++ac) bf[ac] = load_b_frag(Wih, n0 + ac * 16, k0, l15, half);
#pragma unroll
    for (int ar = 0; ar < 2; ++ar)
#pragma unroll
      for (int ac = 0; ac < 2; ++ac) acc[ar][ac] = wmma_bf16(af[ar], bf[ac], acc[ar][ac]);
  }
#pragma unroll
  for (int ar = 0; ar < 2; ++ar)
#pragma unroll
    for (int ac = 0; ac < 2; ++ac) {
      const int col = n0 + ac * 16 + l15;  // C/D layout: lane%16 = N
      const float bias = bih[col] + bhh[col];
#pragma unroll
      for (int v = 0; v < 8; ++v) {
        const int row = r0 + ar * 16 + v + 8 * half;   // VGPR v: M = v + 8*(lane/16)
        XW[(size_t)row * HH + col] = acc[ar][ac][v] + bias;
      }
    }
}

// ---------- device-scope barrier for the persistent scan ----------
__global__ void init_barrier(unsigned int* bar) {
  if (threadIdx.x == 0) bar[0] = 0;
}

__device__ __forceinline__ void grid_barrier(unsigned int* bar, unsigned int target) {
  __syncthreads();
  if (threadIdx.x == 0) {
    __builtin_amdgcn_fence(__ATOMIC_RELEASE, "agent");   // flush WGP$ (global_wb)
    __hip_atomic_fetch_add(bar, 1u, __ATOMIC_RELAXED, __HIP_MEMORY_SCOPE_AGENT);
    while (__hip_atomic_load(bar, __ATOMIC_RELAXED, __HIP_MEMORY_SCOPE_AGENT) < target)
      __builtin_amdgcn_s_sleep(2);
    __builtin_amdgcn_fence(__ATOMIC_ACQUIRE, "agent");   // invalidate WGP$ (global_inv)
  }
  __syncthreads();
}

// ---------- persistent recurrent scan ----------
// 32 WGs (4 row-blocks x 8 col-blocks of a 256x512 step-GEMM), 8 waves each.
// Wave tile 16x32 (2 accums). Whh slice (512x32 bf16) lives in VGPRs for all 256 steps.
// SEQ doubles as recurrent state: step t reads SEQ[t-1], writes SEQ[t].
// sched_barrier(0) pins all 32 A-loads of a step ahead of the WMMA chain so the
// step pays ~one L2 round trip instead of sixteen.
__global__ __launch_bounds__(256, 1) void rnn_scan(const float* __restrict__ XW,
                                                   unsigned short* __restrict__ seq,
                                                   const unsigned short* __restrict__ Whh,
                                                   unsigned int* __restrict__ bar) {
  const int wave = threadIdx.x >> 5, lane = threadIdx.x & 31;
  const int l15 = lane & 15, half = lane >> 4;
  const int wg = blockIdx.x;                      // 0..31
  const int r0 = (wg & 3) * 64 + (wave & 3) * 16; // batch rows
  const int n0 = (wg >> 2) * 64 + (wave >> 2) * 32;
  const int c0 = n0 + l15;

  // Register-resident B operand: 16 K-slices x 2 column tiles = 256 VGPRs
  v16bf breg[16][2];
#pragma unroll
  for (int ki = 0; ki < 16; ++ki) {
    breg[ki][0] = load_b_frag(Whh, n0, ki * 32, l15, half);
    breg[ki][1] = load_b_frag(Whh, n0 + 16, ki * 32, l15, half);
  }

  // Prefetch XW for t = 0
  float xn0[8], xn1[8];
#pragma unroll
  for (int v = 0; v < 8; ++v) {
    const int row = r0 + v + 8 * half;
    xn0[v] = XW[(size_t)row * HH + c0];
    xn1[v] = XW[(size_t)row * HH + c0 + 16];
  }

  for (int t = 0; t < TT; ++t) {
    v8f acc0 = {}, acc1 = {};
    if (t > 0) {
      const unsigned short* hprev =
          seq + (size_t)(t - 1) * BB * HH + (size_t)(r0 + l15) * HH;
      // Batch-issue all 32 b128 A-loads...
      v16bf afr[16];
#pragma unroll
      for (int ki = 0; ki < 16; ++ki) afr[ki] = load_a_bf16(hprev, ki * 32, half);
      // ...and forbid the scheduler from sinking any of them below this point.
      __builtin_amdgcn_sched_barrier(0);
#pragma unroll
      for (int ki = 0; ki < 16; ++ki) {
        acc0 = wmma_bf16(afr[ki], breg[ki][0], acc0);
        acc1 = wmma_bf16(afr[ki], breg[ki][1], acc1);
      }
    }
    unsigned short* hout = seq + (size_t)t * BB * HH;
#pragma unroll
    for (int v = 0; v < 8; ++v) {
      const int row = r0 + v + 8 * half;
      float h0 = tanhf(acc0[v] + xn0[v]);
      float h1 = tanhf(acc1[v] + xn1[v]);
      hout[(size_t)row * HH + c0] = f2bf(h0);
      hout[(size_t)row * HH + c0 + 16] = f2bf(h1);
    }
    // Prefetch next step's XW before the barrier: L2 latency hides in the spin.
    if (t + 1 < TT) {
      const float* xwt = XW + (size_t)(t + 1) * BB * HH;
#pragma unroll
      for (int v = 0; v < 8; ++v) {
        const int row = r0 + v + 8 * half;
        xn0[v] = xwt[(size_t)row * HH + c0];
        xn1[v] = xwt[(size_t)row * HH + c0 + 16];
      }
      __builtin_amdgcn_sched_barrier(0);  // keep these issued before the spin
    }
    grid_barrier(bar, 32u * (unsigned)(t + 1));
  }
}

// ---------- final FC: out = h_{T-1} @ fc_w^T + fc_b, [256 x 512], K=512 ----------
__global__ __launch_bounds__(256) void fc_gemm(const unsigned short* __restrict__ Hlast,
                                               const unsigned short* __restrict__ Wfc,
                                               const float* __restrict__ bfc,
                                               float* __restrict__ out) {
  const int wave = threadIdx.x >> 5, lane = threadIdx.x & 31;
  const int l15 = lane & 15, half = lane >> 4;
  const int W = blockIdx.x * 8 + wave;  // 0..255 wave tiles (16 row x 16 col tiles)
  const int r0 = (W & 15) * 16;
  const int n0 = (W >> 4) * 32;
  v8f acc0 = {}, acc1 = {};
  for (int ki = 0; ki < 16; ++ki) {
    v16bf a = load_a_bf16(Hlast + (size_t)(r0 + l15) * HH, ki * 32, half);
    v16bf b0 = load_b_frag(Wfc, n0, ki * 32, l15, half);
    v16bf b1 = load_b_frag(Wfc, n0 + 16, ki * 32, l15, half);
    acc0 = wmma_bf16(a, b0, acc0);
    acc1 = wmma_bf16(a, b1, acc1);
  }
#pragma unroll
  for (int v = 0; v < 8; ++v) {
    const int row = r0 + v + 8 * half;
    const int c = n0 + l15;
    out[(size_t)row * HH + c] = acc0[v] + bfc[c];
    out[(size_t)row * HH + c + 16] = acc1[v] + bfc[c + 16];
  }
}

// ---------- host launcher ----------
extern "C" void kernel_launch(void* const* d_in, const int* in_sizes, int n_in,
                              void* d_out, int out_size, void* d_ws, size_t ws_size,
                              hipStream_t stream) {
  (void)in_sizes; (void)n_in; (void)out_size; (void)ws_size;
  const float* x      = (const float*)d_in[0];
  const float* Wih[3] = {(const float*)d_in[1], (const float*)d_in[5], (const float*)d_in[9]};
  const float* Whh[3] = {(const float*)d_in[2], (const float*)d_in[6], (const float*)d_in[10]};
  const float* bih[3] = {(const float*)d_in[3], (const float*)d_in[7], (const float*)d_in[11]};
  const float* bhh[3] = {(const float*)d_in[4], (const float*)d_in[8], (const float*)d_in[12]};
  const float* fcw = (const float*)d_in[13];
  const float* fcb = (const float*)d_in[14];
  float* out = (float*)d_out;

  // Workspace layout
  const size_t SEQ_ELEMS = (size_t)TT * BB * HH;           // 33,554,432
  char* ws = (char*)d_ws;
  unsigned int* bar   = (unsigned int*)ws;                 // 256 B
  float* XW           = (float*)(ws + 256);                // 128 MiB fp32
  unsigned short* SEQ = (unsigned short*)(ws + 256 + SEQ_ELEMS * sizeof(float));
  unsigned short* WB  = (unsigned short*)((char*)SEQ + SEQ_ELEMS * sizeof(unsigned short));
  const size_t WMAT = (size_t)HH * HH;                     // 262144 elems per matrix
  unsigned short *wihb[3], *whhb[3];
  for (int l = 0; l < 3; ++l) { wihb[l] = WB + (size_t)(2 * l) * WMAT; whhb[l] = WB + (size_t)(2 * l + 1) * WMAT; }
  unsigned short* fcwb = WB + 6 * WMAT;

  // 1) weights -> bf16
  dim3 cb(256), cg((unsigned)((WMAT + 255) / 256));
  for (int l = 0; l < 3; ++l) {
    cvt_f32_to_bf16<<<cg, cb, 0, stream>>>(Wih[l], wihb[l], (int)WMAT);
    cvt_f32_to_bf16<<<cg, cb, 0, stream>>>(Whh[l], whhb[l], (int)WMAT);
  }
  cvt_f32_to_bf16<<<cg, cb, 0, stream>>>(fcw, fcwb, (int)WMAT);

  // 2) three layers: batched input GEMM, then persistent scan
  for (int l = 0; l < 3; ++l) {
    if (l == 0)
      xw_gemm<true><<<dim3(512, 8), 256, 0, stream>>>((const void*)x, wihb[0], bih[0], bhh[0], XW);
    else
      xw_gemm<false><<<dim3(512, 8), 256, 0, stream>>>((const void*)SEQ, wihb[l], bih[l], bhh[l], XW);
    init_barrier<<<1, 64, 0, stream>>>(bar);
    rnn_scan<<<32, 256, 0, stream>>>(XW, SEQ, whhb[l], bar);
  }

  // 3) final FC from h at t = T-1
  fc_gemm<<<32, 256, 0, stream>>>(SEQ + (size_t)(TT - 1) * BB * HH, fcwb, fcb, out);
}
